// ScaledDotProductAttention_79104707658197
// MI455X (gfx1250) — 8-bit matrix operand order repaired and checked by reading, compiled, not run on hardware
//
#include <hip/hip_runtime.h>
#include <hip/hip_bf16.h>

typedef int v8i __attribute__((ext_vector_type(8)));

#define NB 8
#define NW 1024
#define NH 8
#define ND 64
#define NROWS (NB * NW * NH)
#define CTX_ELEMS (NB * NW * NH * ND)
#define ATT_ELEMS (NB * NH * NW * NW)

struct Stats {
  unsigned mn[3], mx[3];
  int s_min, s_max;
  unsigned att_max;
  int c_min, c_max;
};
struct Params {
  float scale[6];
  int   zp[6];
  int   qmin, qmax;
  float qminf, qmaxf;
};

__device__ __forceinline__ unsigned enc_key(float f) {
  unsigned b = __float_as_uint(f);
  return (b & 0x80000000u) ? ~b : (b | 0x80000000u);
}
__device__ __forceinline__ float dec_key(unsigned k) {
  unsigned b = (k & 0x80000000u) ? (k & 0x7fffffffu) : ~k;
  return __uint_as_float(b);
}

__global__ void init_kernel(Stats* st) {
  if (threadIdx.x == 0) {
    st->mn[0] = st->mn[1] = st->mn[2] = 0xFFFFFFFFu;
    st->mx[0] = st->mx[1] = st->mx[2] = 0u;
    st->s_min = 0x7fffffff; st->s_max = (int)0x80000000;
    st->att_max = 0u;
    st->c_min = 0x7fffffff; st->c_max = (int)0x80000000;
  }
}

__global__ void minmax_kernel(const float* __restrict__ x, unsigned long long n,
                              unsigned* mn_slot, unsigned* mx_slot) {
  unsigned long long i0 = ((unsigned long long)blockIdx.x * blockDim.x + threadIdx.x) * 4ull;
  unsigned long long stride = (unsigned long long)gridDim.x * blockDim.x * 4ull;
  float mn = __uint_as_float(0x7f800000u), mx = -mn;
  for (unsigned long long i = i0; i < n; i += stride) {
    float4 v = *(const float4*)(x + i);
    mn = fminf(mn, fminf(fminf(v.x, v.y), fminf(v.z, v.w)));
    mx = fmaxf(mx, fmaxf(fmaxf(v.x, v.y), fmaxf(v.z, v.w)));
  }
  unsigned kmn = enc_key(mn), kmx = enc_key(mx);
  for (int off = 16; off; off >>= 1) {
    kmn = min(kmn, (unsigned)__shfl_xor(kmn, off));
    kmx = max(kmx, (unsigned)__shfl_xor(kmx, off));
  }
  if ((threadIdx.x & 31) == 0) { atomicMin(mn_slot, kmn); atomicMax(mx_slot, kmx); }
}

__global__ void params_kernel(const int* __restrict__ bits_ptr, Stats* st, Params* p, int stage) {
  if (threadIdx.x != 0 || blockIdx.x != 0) return;
  if (stage == 0) {
    int bits = bits_ptr[0];
    int qmin = -(1 << (bits - 1)), qmax = (1 << (bits - 1)) - 1;
    p->qmin = qmin; p->qmax = qmax;
    p->qminf = (float)qmin; p->qmaxf = (float)qmax;
    float range = (float)(qmax - qmin);
    for (int t = 0; t < 3; ++t) {
      float mn = fminf(dec_key(st->mn[t]), 0.0f);
      float mx = fmaxf(dec_key(st->mx[t]), 0.0f);
      float sc = fmaxf((mx - mn) / range, 1e-8f);
      float zp = fminf(fmaxf(rintf((float)qmin - mn / sc), (float)qmin), (float)qmax);
      p->scale[t] = sc; p->zp[t] = (int)zp;
    }
  } else {
    float range = (float)(p->qmax - p->qmin);
    float mn, mx; int idx;
    if (stage == 1) {
      float s = p->scale[0] * p->scale[1] * 0.125f;
      mn = fminf(s * (float)st->s_min, 0.0f);
      mx = fmaxf(s * (float)st->s_max, 0.0f);
      idx = 3;
    } else if (stage == 2) {
      mn = 0.0f;
      mx = fmaxf(__uint_as_float(st->att_max), 0.0f);
      idx = 4;
    } else {
      float sav = p->scale[4] * p->scale[2];
      mn = fminf(sav * (float)st->c_min, 0.0f);
      mx = fmaxf(sav * (float)st->c_max, 0.0f);
      idx = 5;
    }
    float sc = fmaxf((mx - mn) / range, 1e-8f);
    float zp = fminf(fmaxf(rintf(p->qminf - mn / sc), p->qminf), p->qmaxf);
    p->scale[idx] = sc; p->zp[idx] = (int)zp;
  }
}

__global__ void quantize_qk_kernel(const float* __restrict__ x, signed char* __restrict__ xq,
                                   int* __restrict__ rs, const Params* __restrict__ p, int which) {
  const int lane = threadIdx.x & 31, wave = threadIdx.x >> 5;
  const int row = blockIdx.x * 8 + wave;
  const float sc = p->scale[which], zpf = (float)p->zp[which];
  const float qmn = p->qminf, qmx = p->qmaxf;
  const float* xr = x + (unsigned long long)row * 64ull;
  int q0 = (int)fminf(fmaxf(rintf(xr[lane] / sc) + zpf, qmn), qmx);
  int q1 = (int)fminf(fmaxf(rintf(xr[lane + 32] / sc) + zpf, qmn), qmx);
  signed char* o = xq + (unsigned long long)row * 64ull;
  o[lane] = (signed char)q0;
  o[lane + 32] = (signed char)q1;
  int s = q0 + q1;
  for (int off = 16; off; off >>= 1) s += __shfl_xor(s, off);
  if (lane == 0) rs[row] = s;
}

__global__ void quantize_v_kernel(const float* __restrict__ v, signed char* __restrict__ vt,
                                  const Params* __restrict__ p) {
  __shared__ signed char tile[64][68];
  const int blk = blockIdx.x;
  const int kb = blk & 15, h = (blk >> 4) & 7, b = blk >> 7;
  const int k0 = kb * 64;
  const float sc = p->scale[2], zpf = (float)p->zp[2];
  const float qmn = p->qminf, qmx = p->qmaxf;
  #pragma unroll
  for (int i = 0; i < 16; ++i) {
    int e = threadIdx.x + i * 256;
    int kk = e >> 6, d = e & 63;
    float x = v[((unsigned long long)((b * NW + k0 + kk) * NH + h)) * 64ull + d];
    int qv = (int)fminf(fmaxf(rintf(x / sc) + zpf, qmn), qmx);
    tile[kk][d] = (signed char)qv;
  }
  __syncthreads();
  #pragma unroll
  for (int i = 0; i < 4; ++i) {
    int e = threadIdx.x + i * 256;
    int d = e >> 4, kk = (e & 15) * 4;
    char4 o;
    o.x = tile[kk + 0][d]; o.y = tile[kk + 1][d];
    o.z = tile[kk + 2][d]; o.w = tile[kk + 3][d];
    *(char4*)(vt + ((unsigned long long)((b * NH + h) * 64 + d)) * 1024ull + k0 + kk) = o;
  }
}

__global__ void scores_kernel(const signed char* __restrict__ qi, const signed char* __restrict__ ki,
                              const int* __restrict__ rq, const int* __restrict__ rk,
                              const Params* __restrict__ p, Stats* st, int* __restrict__ s_out) {
  const int lane = threadIdx.x & 31, wave = threadIdx.x >> 5;
  const int gw = blockIdx.x * 8 + wave;
  const int ng = gw & 15, mt = (gw >> 4) & 63, bh = gw >> 10;
  const int b = bh >> 3, h = bh & 7;
  const int m0 = mt * 16, n0 = ng * 64;
  const int lo = lane & 15, hi = lane >> 4;
  const int zq = p->zp[0], zk = p->zp[1];
  const int cc = 64 * zq * zk;

  const signed char* arow = qi + ((unsigned long long)(((b * NW) + m0 + lo) * NH + h)) * 64ull + hi * 16;
  int4 a0 = *(const int4*)(arow);
  int4 a1 = *(const int4*)(arow + 32);
  v8i A = {a0.x, a0.y, a0.z, a0.w, a1.x, a1.y, a1.z, a1.w};

  int rqv[8];
  #pragma unroll
  for (int r = 0; r < 8; ++r)
    rqv[r] = rq[((b * NW) + m0 + r + 8 * hi) * NH + h];

  int smin = 0x7fffffff, smax = (int)0x80000000;
  #pragma unroll
  for (int j = 0; j < 4; ++j) {
    const int n = n0 + j * 16 + lo;
    const signed char* brow = ki + ((unsigned long long)(((b * NW) + n) * NH + h)) * 64ull + hi * 16;
    int4 b0 = *(const int4*)(brow);
    int4 b1 = *(const int4*)(brow + 32);
    v8i Bf = {b0.x, b0.y, b0.z, b0.w, b1.x, b1.y, b1.z, b1.w};
    v8i cz = {0, 0, 0, 0, 0, 0, 0, 0};
    v8i d = __builtin_amdgcn_wmma_i32_16x16x64_iu8(true, A, true, Bf, cz, false, false);
    const int rkn = rk[((b * NW) + n) * NH + h];
    #pragma unroll
    for (int r = 0; r < 8; ++r) {
      int sv = d[r] - zk * rqv[r] - zq * rkn + cc;
      smin = min(smin, sv); smax = max(smax, sv);
      s_out[((unsigned long long)(bh * NW + m0 + r + 8 * hi)) * 1024ull + n] = sv;
    }
  }
  for (int off = 16; off; off >>= 1) {
    smin = min(smin, __shfl_xor(smin, off));
    smax = max(smax, __shfl_xor(smax, off));
  }
  if (lane == 0) { atomicMin(&st->s_min, smin); atomicMax(&st->s_max, smax); }
}

__global__ void softmax_kernel(float* __restrict__ att, const Params* __restrict__ p, Stats* st) {
  const int row = blockIdx.x;
  const int tid = threadIdx.x;
  const float s = p->scale[0] * p->scale[1] * 0.125f;
  const float ss = p->scale[3], zs = (float)p->zp[3];
  const float qmn = p->qminf, qmx = p->qmaxf;
  const int* sint = (const int*)att + (unsigned long long)row * 1024ull;
  int4 v = *(const int4*)(sint + tid * 4);
  float x[4];
  int raw[4] = {v.x, v.y, v.z, v.w};
  #pragma unroll
  for (int i = 0; i < 4; ++i) {
    float f = s * (float)raw[i];
    float q = fminf(fmaxf(rintf(f / ss) + zs, qmn), qmx);
    x[i] = (q - zs) * ss;
  }
  float m = fmaxf(fmaxf(x[0], x[1]), fmaxf(x[2], x[3]));
  for (int off = 16; off; off >>= 1) m = fmaxf(m, __shfl_xor(m, off));
  __shared__ float redmax[8], redsum[8], bc[2];
  if ((tid & 31) == 0) redmax[tid >> 5] = m;
  __syncthreads();
  if (tid == 0) {
    float mm = redmax[0];
    for (int i = 1; i < 8; ++i) mm = fmaxf(mm, redmax[i]);
    bc[0] = mm;
  }
  __syncthreads();
  float rmax = bc[0];
  float e[4], ssum = 0.0f;
  #pragma unroll
  for (int i = 0; i < 4; ++i) { e[i] = expf(x[i] - rmax); ssum += e[i]; }
  for (int off = 16; off; off >>= 1) ssum += __shfl_xor(ssum, off);
  if ((tid & 31) == 0) redsum[tid >> 5] = ssum;
  __syncthreads();
  if (tid == 0) {
    float t = redsum[0];
    for (int i = 1; i < 8; ++i) t += redsum[i];
    bc[1] = t;
  }
  __syncthreads();
  float inv = 1.0f / bc[1];
  float4 o = {e[0] * inv, e[1] * inv, e[2] * inv, e[3] * inv};
  *(float4*)(att + (unsigned long long)row * 1024ull + tid * 4) = o;
  if (tid == 0) atomicMax(&st->att_max, __float_as_uint(inv));
}

__global__ void fq_kernel(float* __restrict__ x, const Params* __restrict__ p, int idx) {
  unsigned long long i = ((unsigned long long)blockIdx.x * 256ull + threadIdx.x) * 4ull;
  const float sc = p->scale[idx], zpf = (float)p->zp[idx];
  const float qmn = p->qminf, qmx = p->qmaxf;
  float4 v = *(float4*)(x + i);
  float4 o;
  o.x = (fminf(fmaxf(rintf(v.x / sc) + zpf, qmn), qmx) - zpf) * sc;
  o.y = (fminf(fmaxf(rintf(v.y / sc) + zpf, qmn), qmx) - zpf) * sc;
  o.z = (fminf(fmaxf(rintf(v.z / sc) + zpf, qmn), qmx) - zpf) * sc;
  o.w = (fminf(fmaxf(rintf(v.w / sc) + zpf, qmn), qmx) - zpf) * sc;
  *(float4*)(x + i) = o;
}

__global__ void context_kernel(const signed char* __restrict__ vt, const float* __restrict__ att,
                               const Params* __restrict__ p, Stats* st, float* __restrict__ ctx) {
  const int lane = threadIdx.x & 31, wave = threadIdx.x >> 5;
  const int gw = blockIdx.x * 8 + wave;
  const int nt = gw & 63, bh = gw >> 6;
  const int b = bh >> 3, h = bh & 7;
  const int wq0 = nt * 16, lo = lane & 15, hi = lane >> 4;
  const float inv_sa = 1.0f / p->scale[4];
  const int zv = p->zp[2];

  v8i zero = {0, 0, 0, 0, 0, 0, 0, 0};
  v8i c[4] = {zero, zero, zero, zero};
  int bsum = 0;
  const float* brow = att + ((unsigned long long)(bh * NW + wq0 + lo)) * 1024ull;

  for (int k0 = 0; k0 < 1024; k0 += 64) {
    int bb[8];
    const float* bp = brow + k0 + hi * 16;
    #pragma unroll
    for (int g = 0; g < 2; ++g) {
      #pragma unroll
      for (int q4 = 0; q4 < 4; ++q4) {
        float4 f = *(const float4*)(bp + g * 32 + q4 * 4);
        int u0 = __float2int_rn(f.x * inv_sa) & 255;
        int u1 = __float2int_rn(f.y * inv_sa) & 255;
        int u2 = __float2int_rn(f.z * inv_sa) & 255;
        int u3 = __float2int_rn(f.w * inv_sa) & 255;
        bsum += u0 + u1 + u2 + u3;
        bb[g * 4 + q4] = u0 | (u1 << 8) | (u2 << 16) | (u3 << 24);
      }
    }
    v8i Bf = {bb[0], bb[1], bb[2], bb[3], bb[4], bb[5], bb[6], bb[7]};
    #pragma unroll
    for (int i = 0; i < 4; ++i) {
      const signed char* arow = vt + ((unsigned long long)(bh * 64 + i * 16 + lo)) * 1024ull + k0 + hi * 16;
      int4 a0 = *(const int4*)(arow);
      int4 a1 = *(const int4*)(arow + 32);
      v8i A = {a0.x, a0.y, a0.z, a0.w, a1.x, a1.y, a1.z, a1.w};
      c[i] = __builtin_amdgcn_wmma_i32_16x16x64_iu8(true, A, false, Bf, c[i], false, false);
    }
  }
  bsum += __shfl_xor(bsum, 16);
  const float sav = p->scale[4] * p->scale[2];
  int cmin = 0x7fffffff, cmax = (int)0x80000000;
  const unsigned long long obase = ((unsigned long long)((b * NW + wq0 + lo) * NH + h)) * 64ull;
  #pragma unroll
  for (int i = 0; i < 4; ++i) {
    #pragma unroll
    for (int r = 0; r < 8; ++r) {
      int cv = c[i][r] - zv * bsum;
      cmin = min(cmin, cv); cmax = max(cmax, cv);
      ctx[obase + i * 16 + r + 8 * hi] = sav * (float)cv;
    }
  }
  for (int off = 16; off; off >>= 1) {
    cmin = min(cmin, __shfl_xor(cmin, off));
    cmax = max(cmax, __shfl_xor(cmax, off));
  }
  if (lane == 0) { atomicMin(&st->c_min, cmin); atomicMax(&st->c_max, cmax); }
}

extern "C" void kernel_launch(void* const* d_in, const int* in_sizes, int n_in,
                              void* d_out, int out_size, void* d_ws, size_t ws_size,
                              hipStream_t stream) {
  (void)in_sizes; (void)n_in; (void)out_size; (void)ws_size;
  const float* q = (const float*)d_in[0];
  const float* k = (const float*)d_in[1];
  const float* v = (const float*)d_in[2];
  const int* bits = (const int*)d_in[3];

  float* ctx_out = (float*)d_out;
  float* att_out = (float*)d_out + (size_t)CTX_ELEMS;

  char* w = (char*)d_ws;
  Stats* st = (Stats*)w;
  Params* p = (Params*)(w + 256);
  int* rq = (int*)(w + (1 << 20));
  int* rk = (int*)(w + (1 << 20) + (1 << 18));
  signed char* qi = (signed char*)(w + (2 << 20));
  signed char* ki = (signed char*)(w + (6 << 20));
  signed char* vt = (signed char*)(w + (10 << 20));

  const unsigned long long N = (unsigned long long)CTX_ELEMS;

  init_kernel<<<1, 32, 0, stream>>>(st);
  minmax_kernel<<<1024, 256, 0, stream>>>(q, N, &st->mn[0], &st->mx[0]);
  minmax_kernel<<<1024, 256, 0, stream>>>(k, N, &st->mn[1], &st->mx[1]);
  minmax_kernel<<<1024, 256, 0, stream>>>(v, N, &st->mn[2], &st->mx[2]);
  params_kernel<<<1, 32, 0, stream>>>(bits, st, p, 0);

  quantize_qk_kernel<<<NROWS / 8, 256, 0, stream>>>(q, qi, rq, p, 0);
  quantize_qk_kernel<<<NROWS / 8, 256, 0, stream>>>(k, ki, rk, p, 1);
  quantize_v_kernel<<<1024, 256, 0, stream>>>(v, vt, p);

  scores_kernel<<<8192, 256, 0, stream>>>(qi, ki, rq, rk, p, st, (int*)att_out);
  params_kernel<<<1, 32, 0, stream>>>(bits, st, p, 1);

  softmax_kernel<<<65536, 256, 0, stream>>>(att_out, p, st);
  params_kernel<<<1, 32, 0, stream>>>(bits, st, p, 2);
  fq_kernel<<<ATT_ELEMS / 1024, 256, 0, stream>>>(att_out, p, 4);

  context_kernel<<<512, 256, 0, stream>>>(vt, att_out, p, st, ctx_out);
  params_kernel<<<1, 32, 0, stream>>>(bits, st, p, 3);
  fq_kernel<<<CTX_ELEMS / 1024, 256, 0, stream>>>(ctx_out, p, 5);
}
